// RetrievalBlock_5866925326436
// MI455X (gfx1250) — compile-verified, matrix operand order checked by reading, not run on hardware
//
#include <hip/hip_runtime.h>

#define HIDDEN   1024
#define KEY_DIM  512
#define VALUE_DIM 1024
#define N_MEM    32768
#define TOKENS   4096
#define CHUNK    256
#define NWAVE    8

typedef __attribute__((ext_vector_type(16))) __bf16 v16bf;
typedef __attribute__((ext_vector_type(8)))  float  v8f;
typedef __attribute__((ext_vector_type(4)))  unsigned int u32x4;
typedef __attribute__((ext_vector_type(8)))  float  f32x8;
typedef __attribute__((ext_vector_type(8)))  unsigned short u16x8;

union FragU { v16bf v; u32x4 q[2]; };

static __device__ __forceinline__ unsigned short f2bf(float f) {
    unsigned u = __float_as_uint(f);
    unsigned r = u + 0x7FFFu + ((u >> 16) & 1u);
    return (unsigned short)(r >> 16);
}

// 16x32 bf16 WMMA operand fragment (A layout; B identical with "row" =
// source row = operand column). ISA 7.12.2: lanes 0-15 hold K
// {kb..kb+7, kb+16..kb+23}, lanes 16-31 hold K {kb+8..kb+15, kb+24..kb+31}.
static __device__ __forceinline__ v16bf frag_ld(const unsigned short* base,
                                                long stride, int row0, int kbase) {
    int lane = threadIdx.x & 31;
    int r = lane & 15, h = lane >> 4;
    const unsigned short* p = base + (long)(row0 + r) * stride + kbase + h * 8;
    FragU f;
    f.q[0] = *(const u32x4*)(p);
    f.q[1] = *(const u32x4*)(p + 16);
    return f.v;
}

static __device__ __forceinline__ v8f wmma_bf16(v16bf a, v16bf b, v8f c) {
    return __builtin_amdgcn_wmma_f32_16x16x32_bf16(false, a, false, b,
                                                   (short)0, c, false, false);
}

static __device__ __forceinline__ float redmax16(float v) {
    for (int m = 1; m <= 8; m <<= 1) v = fmaxf(v, __shfl_xor(v, m, 32));
    return v;
}
static __device__ __forceinline__ float redsum16(float v) {
    for (int m = 1; m <= 8; m <<= 1) v += __shfl_xor(v, m, 32);
    return v;
}

// ---------------- conversion kernels ----------------
__global__ void k_cvt_bf16(const float* __restrict__ src,
                           unsigned short* __restrict__ dst, size_t n8) {
    // n8 = n/8; all our sizes are multiples of 8
    const f32x8* s = (const f32x8*)src;
    u16x8* d = (u16x8*)dst;
    for (size_t i = (size_t)blockIdx.x * blockDim.x + threadIdx.x; i < n8;
         i += (size_t)gridDim.x * blockDim.x) {
        f32x8 v = s[i];
        u16x8 o;
        #pragma unroll
        for (int e = 0; e < 8; ++e) o[e] = f2bf(v[e]);
        d[i] = o;
    }
}

__global__ void k_cvt_transpose(const float* __restrict__ src,
                                unsigned short* __restrict__ dst,
                                int rows, int cols) {
    size_t n = (size_t)rows * cols;
    for (size_t i = (size_t)blockIdx.x * blockDim.x + threadIdx.x; i < n;
         i += (size_t)gridDim.x * blockDim.x) {
        size_t r = i / cols, c = i - r * cols;
        dst[c * (size_t)rows + r] = f2bf(src[i]);
    }
}

// ---------------- Q = hs * Wq^T + bq ----------------
__global__ void k_queries(const unsigned short* __restrict__ hid,
                          const unsigned short* __restrict__ wq,
                          const float* __restrict__ bq,
                          unsigned short* __restrict__ qout) {
    int rowbase = blockIdx.x * 16;
    int wv = threadIdx.x >> 5, lane = threadIdx.x & 31;
    int h = lane >> 4, nl = lane & 15;
    int n0 = wv * 64;

    v8f acc[4] = {};
    // double-buffered operand pipeline over the K dimension
    v16bf a = frag_ld(hid, HIDDEN, rowbase, 0);
    v16bf b[4];
    #pragma unroll
    for (int t = 0; t < 4; ++t) b[t] = frag_ld(wq, HIDDEN, n0 + t * 16, 0);

    #pragma unroll 2
    for (int ks = 0; ks < HIDDEN / 32; ++ks) {
        int ksn = (ks + 1) & (HIDDEN / 32 - 1);
        v16bf an = frag_ld(hid, HIDDEN, rowbase, ksn * 32);
        v16bf bn[4];
        #pragma unroll
        for (int t = 0; t < 4; ++t) bn[t] = frag_ld(wq, HIDDEN, n0 + t * 16, ksn * 32);
        #pragma unroll
        for (int t = 0; t < 4; ++t) acc[t] = wmma_bf16(a, b[t], acc[t]);
        a = an;
        #pragma unroll
        for (int t = 0; t < 4; ++t) b[t] = bn[t];
    }

    #pragma unroll
    for (int t = 0; t < 4; ++t) {
        int col = n0 + t * 16 + nl;
        float bv = bq[col];
        #pragma unroll
        for (int i = 0; i < 8; ++i) {
            int row = h * 8 + i;
            qout[(size_t)(rowbase + row) * KEY_DIM + col] = f2bf(acc[t][i] + bv);
        }
    }
}

// ------- fused scores + online softmax + retrieval (flash style) -------
__global__ void k_retrieve(const unsigned short* __restrict__ qbf,
                           const unsigned short* __restrict__ keys,
                           const unsigned short* __restrict__ valT,
                           unsigned short* __restrict__ retr) {
    __shared__ unsigned short sQ[16][KEY_DIM];   // 16 KB
    __shared__ unsigned short sP[16][CHUNK];     // 8 KB
    __shared__ float sMaxP[NWAVE][16];
    __shared__ float sSumP[NWAVE][16];
    __shared__ float sM[16], sL[16], sScale[16];

    int rowbase = blockIdx.x * 16;
    int wv = threadIdx.x >> 5, lane = threadIdx.x & 31;
    int h = lane >> 4, nl = lane & 15;
    int cb = wv * 32;

    {   // stage Q tile (16x512 bf16, contiguous) into LDS
        const u32x4* src = (const u32x4*)(qbf + (size_t)rowbase * KEY_DIM);
        u32x4* dst = (u32x4*)&sQ[0][0];
        for (int i = threadIdx.x; i < 16 * KEY_DIM / 8; i += 256) dst[i] = src[i];
    }
    if (threadIdx.x < 16) { sM[threadIdx.x] = -1e30f; sL[threadIdx.x] = 0.0f; }
    __syncthreads();

    v8f O[8] = {};   // 16 x 128 f32 accumulator per wave

    for (int c = 0; c < N_MEM / CHUNK; ++c) {
        int mb = c * CHUNK;

        // prefetch next chunk's key rows / value columns toward this WGP
        if (c + 1 < N_MEM / CHUNK) {
            __builtin_prefetch(keys + (size_t)(mb + CHUNK + cb + nl) * KEY_DIM + h * 256, 0, 1);
            __builtin_prefetch(valT + (size_t)(wv * 128 + lane * 4) * N_MEM + mb + CHUNK, 0, 1);
        }

        // -------- phase 1: scores for this wave's 32 memory columns --------
        v8f s0 = {}, s1 = {};
        v16bf b0 = frag_ld(keys, KEY_DIM, mb + cb, 0);
        v16bf b1 = frag_ld(keys, KEY_DIM, mb + cb + 16, 0);
        #pragma unroll 2
        for (int ks = 0; ks < KEY_DIM / 32; ++ks) {
            int ksn = (ks + 1) & (KEY_DIM / 32 - 1);
            v16bf b0n = frag_ld(keys, KEY_DIM, mb + cb, ksn * 32);
            v16bf b1n = frag_ld(keys, KEY_DIM, mb + cb + 16, ksn * 32);
            v16bf aq = frag_ld(&sQ[0][0], KEY_DIM, 0, ks * 32);
            s0 = wmma_bf16(aq, b0, s0);
            s1 = wmma_bf16(aq, b1, s1);
            b0 = b0n; b1 = b1n;
        }

        // -------- phase 2: online softmax bookkeeping --------
        float mx[8];
        #pragma unroll
        for (int i = 0; i < 8; ++i) mx[i] = redmax16(fmaxf(s0[i], s1[i]));
        if (lane == 0) {
            for (int i = 0; i < 8; ++i) sMaxP[wv][i] = mx[i];
        }
        if (lane == 16) {
            for (int i = 0; i < 8; ++i) sMaxP[wv][8 + i] = mx[i];
        }
        __syncthreads();

        if (threadIdx.x < 16) {
            int r = threadIdx.x;
            float cm = -1e30f;
            for (int w = 0; w < NWAVE; ++w) cm = fmaxf(cm, sMaxP[w][r]);
            float mp = sM[r];
            float mn = fmaxf(mp, cm);
            float sc = __expf(mp - mn);
            sScale[r] = sc;
            sM[r] = mn;
            sL[r] *= sc;
        }
        __syncthreads();

        float mn_l[8], sc_l[8];
        #pragma unroll
        for (int i = 0; i < 8; ++i) { int r = h * 8 + i; mn_l[i] = sM[r]; sc_l[i] = sScale[r]; }
        #pragma unroll
        for (int j = 0; j < 8; ++j) {
            #pragma unroll
            for (int i = 0; i < 8; ++i) O[j][i] *= sc_l[i];
        }

        float rs[8];
        #pragma unroll
        for (int i = 0; i < 8; ++i) {
            float p0 = __expf(s0[i] - mn_l[i]);
            float p1 = __expf(s1[i] - mn_l[i]);
            int row = h * 8 + i;
            sP[row][cb + nl] = f2bf(p0);
            sP[row][cb + 16 + nl] = f2bf(p1);
            rs[i] = redsum16(p0 + p1);
        }
        if (lane == 0) {
            for (int i = 0; i < 8; ++i) sSumP[wv][i] = rs[i];
        }
        if (lane == 16) {
            for (int i = 0; i < 8; ++i) sSumP[wv][8 + i] = rs[i];
        }
        __syncthreads();

        if (threadIdx.x < 16) {
            int r = threadIdx.x;
            float t = 0.0f;
            for (int w = 0; w < NWAVE; ++w) t += sSumP[w][r];
            sL[r] += t;
        }

        // -------- phase 3: O += P(16x256) * V(256x128) for this wave's cols ----
        // hoist all P fragments (LDS) once; batch B loads per output tile so
        // the scheduler can stagger s_wait_loadcnt instead of waiting to zero
        v16bf aP[8];
        #pragma unroll
        for (int kk = 0; kk < CHUNK / 32; ++kk)
            aP[kk] = frag_ld(&sP[0][0], CHUNK, 0, kk * 32);

        for (int j = 0; j < 8; ++j) {
            int vb = wv * 128 + j * 16;
            v16bf bf[8];
            #pragma unroll
            for (int kk = 0; kk < CHUNK / 32; ++kk)
                bf[kk] = frag_ld(valT, N_MEM, vb, mb + kk * 32);
            #pragma unroll
            for (int kk = 0; kk < CHUNK / 32; ++kk)
                O[j] = wmma_bf16(aP[kk], bf[kk], O[j]);
        }
        __syncthreads();
    }

    // -------- finalize: divide by l, write retrieved (bf16) --------
    float inv[8];
    #pragma unroll
    for (int i = 0; i < 8; ++i) inv[i] = 1.0f / sL[h * 8 + i];
    #pragma unroll
    for (int j = 0; j < 8; ++j) {
        int col = wv * 128 + j * 16 + nl;
        #pragma unroll
        for (int i = 0; i < 8; ++i) {
            int row = h * 8 + i;
            retr[(size_t)(rowbase + row) * VALUE_DIM + col] = f2bf(O[j][i] * inv[i]);
        }
    }
}

// ------- update = [hs|retr] * Wg^T + bg; x = hs + update; LayerNorm -------
__global__ void k_update_ln(const unsigned short* __restrict__ hid,
                            const unsigned short* __restrict__ retr,
                            const unsigned short* __restrict__ wg,
                            const float* __restrict__ bg,
                            const float* __restrict__ hs_f32,
                            const float* __restrict__ gamma,
                            const float* __restrict__ beta,
                            float* __restrict__ out) {
    __shared__ float sX[16][HIDDEN];   // 64 KB
    __shared__ float sMu[16], sR[16];

    int rowbase = blockIdx.x * 16;
    int wv = threadIdx.x >> 5, lane = threadIdx.x & 31;
    int h = lane >> 4, nl = lane & 15;
    int n0 = wv * 128;

    v8f acc[8] = {};
    for (int ks = 0; ks < 64; ++ks) {        // K = 2048 = hidden(1024) + retrieved(1024)
        const unsigned short* abase = (ks < 32) ? hid : retr;
        int kb = (ks < 32 ? ks : ks - 32) * 32;
        // batch the A + 8 B loads before the 8 WMMAs for in-iteration overlap
        v16bf a = frag_ld(abase, 1024, rowbase, kb);
        v16bf b[8];
        #pragma unroll
        for (int t = 0; t < 8; ++t)
            b[t] = frag_ld(wg, HIDDEN + VALUE_DIM, n0 + t * 16, ks * 32);
        #pragma unroll
        for (int t = 0; t < 8; ++t) acc[t] = wmma_bf16(a, b[t], acc[t]);
    }

    #pragma unroll
    for (int t = 0; t < 8; ++t) {
        int col = n0 + t * 16 + nl;
        float bv = bg[col];
        #pragma unroll
        for (int i = 0; i < 8; ++i) {
            int row = h * 8 + i;
            sX[row][col] = acc[t][i] + bv +
                           hs_f32[(size_t)(rowbase + row) * HIDDEN + col];
        }
    }
    __syncthreads();

    // LayerNorm: 2 rows per wave
    for (int rr = 0; rr < 2; ++rr) {
        int row = wv * 2 + rr;
        float s = 0.0f, s2 = 0.0f;
        for (int col = lane; col < HIDDEN; col += 32) {
            float x = sX[row][col];
            s += x; s2 += x * x;
        }
        for (int m = 16; m; m >>= 1) { s += __shfl_xor(s, m, 32); s2 += __shfl_xor(s2, m, 32); }
        if (lane == 0) {
            float mu = s * (1.0f / HIDDEN);
            float var = s2 * (1.0f / HIDDEN) - mu * mu;
            sMu[row] = mu;
            sR[row] = rsqrtf(var + 1e-5f);
        }
    }
    __syncthreads();

    for (int idx = threadIdx.x; idx < 16 * HIDDEN; idx += 256) {
        int row = idx >> 10, col = idx & (HIDDEN - 1);
        out[(size_t)(rowbase + row) * HIDDEN + col] =
            (sX[row][col] - sMu[row]) * sR[row] * gamma[col] + beta[col];
    }
}

extern "C" void kernel_launch(void* const* d_in, const int* in_sizes, int n_in,
                              void* d_out, int out_size, void* d_ws, size_t ws_size,
                              hipStream_t stream) {
    const float* hs    = (const float*)d_in[0];   // [4096,1024]
    const float* Wq    = (const float*)d_in[1];   // [512,1024]
    const float* bq    = (const float*)d_in[2];   // [512]
    const float* keysf = (const float*)d_in[3];   // [32768,512]
    const float* valsf = (const float*)d_in[4];   // [32768,1024]
    const float* Wg    = (const float*)d_in[5];   // [1024,2048]
    const float* bg    = (const float*)d_in[6];   // [1024]
    const float* gamma = (const float*)d_in[7];   // [1024]
    const float* beta  = (const float*)d_in[8];   // [1024]
    float* out = (float*)d_out;

    // workspace layout (bf16 as ushort)
    unsigned short* hid_b  = (unsigned short*)d_ws;
    unsigned short* wq_b   = hid_b  + (size_t)TOKENS * HIDDEN;
    unsigned short* keys_b = wq_b   + (size_t)KEY_DIM * HIDDEN;
    unsigned short* valT_b = keys_b + (size_t)N_MEM * KEY_DIM;
    unsigned short* wg_b   = valT_b + (size_t)VALUE_DIM * N_MEM;
    unsigned short* q_b    = wg_b   + (size_t)HIDDEN * (HIDDEN + VALUE_DIM);
    unsigned short* retr_b = q_b    + (size_t)TOKENS * KEY_DIM;

    k_cvt_bf16<<<1024, 256, 0, stream>>>(hs,    hid_b,  (size_t)TOKENS * HIDDEN / 8);
    k_cvt_bf16<<<512,  256, 0, stream>>>(Wq,    wq_b,   (size_t)KEY_DIM * HIDDEN / 8);
    k_cvt_bf16<<<2048, 256, 0, stream>>>(keysf, keys_b, (size_t)N_MEM * KEY_DIM / 8);
    k_cvt_bf16<<<1024, 256, 0, stream>>>(Wg,    wg_b,   (size_t)HIDDEN * (HIDDEN + VALUE_DIM) / 8);
    k_cvt_transpose<<<8192, 256, 0, stream>>>(valsf, valT_b, N_MEM, VALUE_DIM);

    k_queries<<<TOKENS / 16, 256, 0, stream>>>(hid_b, wq_b, bq, q_b);
    k_retrieve<<<TOKENS / 16, 256, 0, stream>>>(q_b, keys_b, valT_b, retr_b);
    k_update_ln<<<TOKENS / 16, 256, 0, stream>>>(hid_b, retr_b, wg_b, bg, hs,
                                                 gamma, beta, out);
}
